// LSTMgc_6485400617148
// MI455X (gfx1250) — matrix operand order repaired and checked by reading, not compiled, not run on hardware
//
#include <hip/hip_runtime.h>
#include <stdint.h>

// Problem constants (fixed by the reference)
#define V_   32
#define LAG_ 1024
#define H_   512
#define M_   1024
#define B_   256
#define H4_  2048      // 4*H
#define VH_  16384     // V*H

// WMMA fragment types (CDNA5: v_wmma_f32_16x16x32_bf16, wave32)
typedef __attribute__((ext_vector_type(16))) __bf16 v16bf;
typedef __attribute__((ext_vector_type(8)))  float  v8f;

union Frag { v16bf v; uint4 q[2]; };   // 8 VGPRs of packed bf16 pairs

// Two fp32 -> packed bf16x2 in 3 VALU ops: round-half-up then byte-select
// via V_PERM_B32 (combined {src0,src1}: src1 = bytes 0-3, src0 = bytes 4-7).
__device__ __forceinline__ uint32_t pack2bf(float lo, float hi) {
  uint32_t a = __float_as_uint(lo) + 0x8000u;
  uint32_t b = __float_as_uint(hi) + 0x8000u;
  return __builtin_amdgcn_perm(b, a, 0x07060302u); // {b3,b2,a3,a2}
}
__device__ __forceinline__ unsigned short f2bf(float f) {   // scalar RNE (epilogue)
  uint32_t x = __float_as_uint(f);
  x += 0x7FFFu + ((x >> 16) & 1u);
  return (unsigned short)(x >> 16);
}
__device__ __forceinline__ float bf2f(unsigned short b) {
  return __uint_as_float((uint32_t)b << 16);
}
__device__ __forceinline__ float fast_sigmoid(float x) {
  x = fminf(fmaxf(x, -30.f), 30.f);
  return 1.0f / (1.0f + __expf(-x));
}
__device__ __forceinline__ float fast_tanh(float x) {
  x = fminf(fmaxf(x, -15.f), 15.f);
  float e = __expf(2.f * x);
  return (e - 1.f) / (e + 1.f);
}

#define KC    32   // K-chunk per LDS stage (one WMMA K-step)
#define PITCH 40   // bf16 elems per LDS row: 80 B -> every fragment b128 is 16B-aligned
#define HT1   64   // h columns per block in kernel 1

// ---------------------------------------------------------------------------
// Kernel 1: gates = x @ W_ih[v].T (i and g slices only); c = sig(i)*tanh(g)*iw
// grid = (H/64, V); block = 256 (8 waves); block holds all 256 batch rows so
// each W_ih slice streams from HBM exactly once. Global->reg double buffering
// hides HBM latency under the 16-WMMA compute block of each K-chunk.
// ---------------------------------------------------------------------------
__global__ __launch_bounds__(256) void k1_lstm_gates(
    const float* __restrict__ x,      // [B, V, LAG]
    const float* __restrict__ Wih,    // [V, 4H, LAG]
    const float* __restrict__ bih,    // [V, 4H]
    const float* __restrict__ bhh,    // [V, 4H]
    const float* __restrict__ impw,   // [V]
    unsigned short* __restrict__ agg) // [B, V*H] bf16 bits
{
  __shared__ unsigned short sA [B_  * PITCH];  // 256 x 32 (+pad) bf16 = 20 KB
  __shared__ unsigned short sBi[HT1 * PITCH];  //  64 x 32 i-gate rows
  __shared__ unsigned short sBg[HT1 * PITCH];  //  64 x 32 g-gate rows

  const int tid  = threadIdx.x;
  const int lane = tid & 31;
  const int wave = tid >> 5;        // 0..7 -> 32-row batch band
  const int half = lane >> 4;       // lane group 0-15 / 16-31
  const int m    = lane & 15;

  const int hBase = blockIdx.x * HT1;
  const int v     = blockIdx.y;

  const float* wv = Wih + (size_t)v * H4_ * LAG_;

  const v8f vzero = {0.f,0.f,0.f,0.f,0.f,0.f,0.f,0.f};
  v8f acc_i[2][4], acc_g[2][4];
#pragma unroll
  for (int ti = 0; ti < 2; ++ti)
#pragma unroll
    for (int tj = 0; tj < 4; ++tj) { acc_i[ti][tj] = vzero; acc_g[ti][tj] = vzero; }

  // Staging maps: A -> thread owns batch row `tid` (32 f32);
  // B -> 2 gate matrices x 64 rows x 2 half-chunks of 16 f32.
  const int bMat = tid >> 7;           // 0 -> i gate, 1 -> g gate
  const int bRow = (tid & 127) >> 1;   // 0..63
  const int bKg  = (tid & 1) * 16;     // 0 / 16

  const float* aSrc = x + (size_t)tid * (V_ * LAG_) + (size_t)v * LAG_;
  const int gateRow = (bMat == 0) ? (hBase + bRow) : (2 * H_ + hBase + bRow);
  const float* bSrc = wv + (size_t)gateRow * LAG_ + bKg;
  unsigned short* sBsel = (bMat == 0) ? sBi : sBg;

  // Preload chunk 0 into registers
  float4 ra[8], rb[4];
#pragma unroll
  for (int j = 0; j < 8; ++j) ra[j] = *(const float4*)(aSrc + 4 * j);
#pragma unroll
  for (int j = 0; j < 4; ++j) rb[j] = *(const float4*)(bSrc + 4 * j);

  for (int kb = 0; kb < LAG_; kb += KC) {
    { // Convert register-resident chunk into LDS (v_perm packing)
      uint32_t* da = (uint32_t*)&sA[tid * PITCH];
#pragma unroll
      for (int j = 0; j < 8; ++j) {
        da[2*j+0] = pack2bf(ra[j].x, ra[j].y);
        da[2*j+1] = pack2bf(ra[j].z, ra[j].w);
      }
      uint32_t* db = (uint32_t*)&sBsel[bRow * PITCH + bKg];
#pragma unroll
      for (int j = 0; j < 4; ++j) {
        db[2*j+0] = pack2bf(rb[j].x, rb[j].y);
        db[2*j+1] = pack2bf(rb[j].z, rb[j].w);
      }
    }
    __syncthreads();   // stores visible to all waves' fragment loads

    // Issue next chunk's global loads; latency hidden under the WMMA block
    if (kb + KC < LAG_) {
      const int kn = kb + KC;
      __builtin_prefetch(bSrc + kn + KC, 0, 3);
#pragma unroll
      for (int j = 0; j < 8; ++j) ra[j] = *(const float4*)(aSrc + kn + 4 * j);
#pragma unroll
      for (int j = 0; j < 4; ++j) rb[j] = *(const float4*)(bSrc + kn + 4 * j);
    }

    // Fragments in contiguous K order, K = half*16 + 0..15, same for A and B (A 16x32, B 32x16, bf16)
    Frag a[2], fbi[4], fbg[4];
#pragma unroll
    for (int ti = 0; ti < 2; ++ti) {
      const unsigned short* p = &sA[(wave * 32 + ti * 16 + m) * PITCH + half * 16];
      a[ti].q[0] = *(const uint4*)(p);        // K = half*16 + 0..7
      a[ti].q[1] = *(const uint4*)(p + 8);    // K = half*16 + 8..15
    }
#pragma unroll
    for (int tj = 0; tj < 4; ++tj) {
      const unsigned short* pi = &sBi[(tj * 16 + m) * PITCH + half * 16];
      fbi[tj].q[0] = *(const uint4*)(pi);     // K = half*16 + 0..7
      fbi[tj].q[1] = *(const uint4*)(pi + 8); // K = half*16 + 8..15
      const unsigned short* pg = &sBg[(tj * 16 + m) * PITCH + half * 16];
      fbg[tj].q[0] = *(const uint4*)(pg);
      fbg[tj].q[1] = *(const uint4*)(pg + 8);
    }
#pragma unroll
    for (int ti = 0; ti < 2; ++ti)
#pragma unroll
      for (int tj = 0; tj < 4; ++tj) {
        acc_i[ti][tj] = __builtin_amdgcn_wmma_f32_16x16x32_bf16(
            false, a[ti].v, false, fbi[tj].v, (short)0, acc_i[ti][tj], false, false);
        acc_g[ti][tj] = __builtin_amdgcn_wmma_f32_16x16x32_bf16(
            false, a[ti].v, false, fbg[tj].v, (short)0, acc_g[ti][tj], false, false);
      }
    __syncthreads();   // fragment loads done before next chunk's LDS stores
  }

  // Epilogue: c = sigmoid(i + bias_i) * tanh(g + bias_g) * imp_w[v] -> agg bf16
  const float iw = impw[v];
#pragma unroll
  for (int tj = 0; tj < 4; ++tj) {
    const int h = hBase + tj * 16 + m;
    const float biasI = bih[v * H4_ + h]          + bhh[v * H4_ + h];
    const float biasG = bih[v * H4_ + 2 * H_ + h] + bhh[v * H4_ + 2 * H_ + h];
#pragma unroll
    for (int ti = 0; ti < 2; ++ti)
#pragma unroll
      for (int r = 0; r < 8; ++r) {              // C layout: M = r + 8*half, N = m
        const int brow = wave * 32 + ti * 16 + half * 8 + r;
        const float gi = fast_sigmoid(acc_i[ti][tj][r] + biasI);
        const float gg = fast_tanh   (acc_g[ti][tj][r] + biasG);
        agg[(size_t)brow * VH_ + v * H_ + h] = f2bf(gi * gg * iw);
      }
  }
}

// ---------------------------------------------------------------------------
// Kernel 2: hid = relu(agg(bf16) @ mlp1_w.T + b1), K = 16384
// grid = (B/128, M/64); block = 256 (waves 4x2, each 32x32), double-buffered.
// ---------------------------------------------------------------------------
#define BT2 128
#define CT2 64

__global__ __launch_bounds__(256) void k2_mlp1(
    const unsigned short* __restrict__ agg, // [B, VH] bf16
    const float* __restrict__ w1,           // [M, VH]
    const float* __restrict__ b1,           // [M]
    unsigned short* __restrict__ hid)       // [B, M] bf16
{
  __shared__ unsigned short sA[BT2 * PITCH];
  __shared__ unsigned short sB[CT2 * PITCH];

  const int tid   = threadIdx.x;
  const int lane  = tid & 31;
  const int wave  = tid >> 5;
  const int waveR = wave & 3;      // 0..3 -> 32-row band
  const int waveC = wave >> 2;     // 0..1 -> 32-col band
  const int half  = lane >> 4;
  const int m     = lane & 15;

  const int bBase = blockIdx.x * BT2;
  const int cBase = blockIdx.y * CT2;

  const v8f vzero = {0.f,0.f,0.f,0.f,0.f,0.f,0.f,0.f};
  v8f acc[2][2];
#pragma unroll
  for (int ti = 0; ti < 2; ++ti)
#pragma unroll
    for (int tj = 0; tj < 2; ++tj) acc[ti][tj] = vzero;

  const int aRow = tid >> 1, aKg = (tid & 1) * 16;  // A: 128 rows x 32 bf16
  const int wRow = tid >> 2, wKg = (tid & 3) * 8;   // B:  64 rows x 32 f32

  const unsigned short* aSrc = agg + (size_t)(bBase + aRow) * VH_ + aKg;
  const float*          wSrc = w1  + (size_t)(cBase + wRow) * VH_ + wKg;

  uint4  rA[2];
  float4 rW[2];
  rA[0] = ((const uint4*)aSrc)[0];
  rA[1] = ((const uint4*)aSrc)[1];
  rW[0] = *(const float4*)(wSrc);
  rW[1] = *(const float4*)(wSrc + 4);

  for (int kb = 0; kb < VH_; kb += KC) {
    { // Stage current register chunk into LDS
      uint4* da = (uint4*)&sA[aRow * PITCH + aKg];
      da[0] = rA[0];
      da[1] = rA[1];
      uint32_t* db = (uint32_t*)&sB[wRow * PITCH + wKg];
      db[0] = pack2bf(rW[0].x, rW[0].y);
      db[1] = pack2bf(rW[0].z, rW[0].w);
      db[2] = pack2bf(rW[1].x, rW[1].y);
      db[3] = pack2bf(rW[1].z, rW[1].w);
    }
    __syncthreads();

    if (kb + KC < VH_) {    // next chunk's loads overlap the WMMA block
      const int kn = kb + KC;
      __builtin_prefetch(wSrc + kn + KC, 0, 3);
      rA[0] = ((const uint4*)(aSrc + kn))[0];
      rA[1] = ((const uint4*)(aSrc + kn))[1];
      rW[0] = *(const float4*)(wSrc + kn);
      rW[1] = *(const float4*)(wSrc + kn + 4);
    }

    Frag a[2], bfr[2];
#pragma unroll
    for (int ti = 0; ti < 2; ++ti) {
      const unsigned short* p = &sA[(waveR * 32 + ti * 16 + m) * PITCH + half * 16];
      a[ti].q[0] = *(const uint4*)(p);
      a[ti].q[1] = *(const uint4*)(p + 8);
    }
#pragma unroll
    for (int tj = 0; tj < 2; ++tj) {
      const unsigned short* p = &sB[(waveC * 32 + tj * 16 + m) * PITCH + half * 16];
      bfr[tj].q[0] = *(const uint4*)(p);
      bfr[tj].q[1] = *(const uint4*)(p + 8);
    }
#pragma unroll
    for (int ti = 0; ti < 2; ++ti)
#pragma unroll
      for (int tj = 0; tj < 2; ++tj)
        acc[ti][tj] = __builtin_amdgcn_wmma_f32_16x16x32_bf16(
            false, a[ti].v, false, bfr[tj].v, (short)0, acc[ti][tj], false, false);
    __syncthreads();
  }

#pragma unroll
  for (int tj = 0; tj < 2; ++tj) {
    const int col  = cBase + waveC * 32 + tj * 16 + m;
    const float bb = b1[col];
#pragma unroll
    for (int ti = 0; ti < 2; ++ti)
#pragma unroll
      for (int r = 0; r < 8; ++r) {
        const int row = bBase + waveR * 32 + ti * 16 + half * 8 + r;
        hid[(size_t)row * M_ + col] = f2bf(fmaxf(acc[ti][tj][r] + bb, 0.f));
      }
  }
}

// ---------------------------------------------------------------------------
// Kernel 3: out[b] = hid[b,:] . mlp2_w[0,:] + mlp2_b[0]   (O = 1)
// ---------------------------------------------------------------------------
__global__ __launch_bounds__(256) void k3_out(
    const unsigned short* __restrict__ hid, // [B, M] bf16
    const float* __restrict__ w2,           // [1, M]
    const float* __restrict__ b2,           // [1]
    float* __restrict__ out)                // [B]
{
  __shared__ float red[256];
  const int b = blockIdx.x;
  const int t = threadIdx.x;
  float s = 0.f;
#pragma unroll
  for (int k = t; k < M_; k += 256)
    s += bf2f(hid[(size_t)b * M_ + k]) * w2[k];
  red[t] = s;
  __syncthreads();
  for (int off = 128; off > 0; off >>= 1) {
    if (t < off) red[t] += red[t + off];
    __syncthreads();
  }
  if (t == 0) out[b] = red[0] + b2[0];
}

// ---------------------------------------------------------------------------
extern "C" void kernel_launch(void* const* d_in, const int* in_sizes, int n_in,
                              void* d_out, int out_size, void* d_ws, size_t ws_size,
                              hipStream_t stream) {
  (void)in_sizes; (void)n_in; (void)out_size; (void)ws_size;
  const float* x    = (const float*)d_in[0];
  const float* Wih  = (const float*)d_in[1];
  // d_in[2] = W_hh is mathematically unused (h0 == 0)
  const float* bih  = (const float*)d_in[3];
  const float* bhh  = (const float*)d_in[4];
  const float* impw = (const float*)d_in[5];
  const float* w1   = (const float*)d_in[6];
  const float* b1   = (const float*)d_in[7];
  const float* w2   = (const float*)d_in[8];
  const float* b2   = (const float*)d_in[9];

  // Workspace layout: agg [B, V*H] bf16 (8 MB), hid [B, M] bf16 (0.5 MB)
  unsigned short* agg = (unsigned short*)d_ws;
  unsigned short* hid = agg + (size_t)B_ * VH_;

  k1_lstm_gates<<<dim3(H_ / HT1, V_), 256, 0, stream>>>(x, Wih, bih, bhh, impw, agg);
  k2_mlp1<<<dim3(B_ / BT2, M_ / CT2), 256, 0, stream>>>(agg, w1, b1, hid);
  k3_out<<<dim3(B_), 256, 0, stream>>>(hid, w2, b2, (float*)d_out);
}
